// MultiHeadAttention_58265526338063
// MI455X (gfx1250) — compile-verified, matrix operand order checked by reading, not run on hardware
//
#include <hip/hip_runtime.h>

// ---------------------------------------------------------------------------
// MultiHeadAttention forward for MI455X (gfx1250, wave32, WMMA)
//   q/k/v proj (f16 WMMA, f32 accum, double-buffered LDS, reg-prefetch)
//   -> RoPE -> flash attention (double-buffered, async-LDS staging)
//   -> out proj
// ---------------------------------------------------------------------------

typedef __attribute__((ext_vector_type(16))) _Float16 v16h;
typedef __attribute__((ext_vector_type(8)))  float    v8f;
typedef unsigned short u16;
typedef unsigned int   u32;

#define B_  4
#define N_  2048
#define D_  1024
#define H_  16
#define HD_ 64
#define BN_ (B_ * N_)
#define SCALE_ 0.125f   // 64^-0.5

// ---- CDNA5 async global->LDS copy (ASYNCcnt) if the toolchain exposes it ----
#if defined(__has_builtin)
#if __has_builtin(__builtin_amdgcn_global_load_async_to_lds_b128) && \
    __has_builtin(__builtin_amdgcn_s_wait_asynccnt)
#define USE_ASYNC_COPY 1
#endif
#endif
#ifndef USE_ASYNC_COPY
#define USE_ASYNC_COPY 0
#endif

// ---------------------------- helpers --------------------------------------

__device__ inline u16 f2h(float f) {
    union { _Float16 h; u16 u; } c;
    c.h = (_Float16)f;
    return c.u;
}

union FragU { uint4 q[2]; v16h v; };

// 16x32 f16 A/B fragment load: per-lane 16 elems as two contiguous 16B chunks.
// lanes 0-15: K = {0..7, 16..23}; lanes 16-31: K = {8..15, 24..31}  (+k0)
__device__ inline v16h load_frag(const u16* row, int k0, int laneHalf) {
    FragU f;
    const uint4* p0 = (const uint4*)(row + k0 + laneHalf * 8);
    const uint4* p1 = (const uint4*)(row + k0 + laneHalf * 8 + 16);
    f.q[0] = *p0;
    f.q[1] = *p1;
    return f.v;
}

__device__ inline v8f wmma16(v16h a, v16h b, v8f c) {
    return __builtin_amdgcn_wmma_f32_16x16x32_f16(
        /*neg_a=*/false, a, /*neg_b=*/false, b,
        /*c_mod=*/(short)0, c, /*reuse_a=*/false, /*reuse_b=*/false);
}

#if USE_ASYNC_COPY
// builtin signature (from hipcc diagnostic): param0 = int4* in global AS
typedef int v4i_vs __attribute__((vector_size(16)));
typedef __attribute__((address_space(1))) v4i_vs* as1_v4i;
typedef __attribute__((address_space(3))) v4i_vs* as3_v4i;

__device__ inline void async_copy_b128(const u16* gsrc, u16* ldst) {
    __builtin_amdgcn_global_load_async_to_lds_b128(
        (as1_v4i)gsrc, (as3_v4i)ldst, 0, 0);
}
#endif

// ------------------------ conversion kernels --------------------------------

__global__ __launch_bounds__(256)
void f32_to_f16_kernel(const float* __restrict__ x, u16* __restrict__ y, int n) {
    int i = (blockIdx.x * 256 + threadIdx.x) * 4;
    if (i < n) {
        float4 f = *(const float4*)(x + i);
        ushort4 o;
        o.x = f2h(f.x); o.y = f2h(f.y); o.z = f2h(f.z); o.w = f2h(f.w);
        *(ushort4*)(y + i) = o;
    }
}

// W[K][N] f32  ->  Wt[N][K] f16   (LDS tile transpose, coalesced both sides)
__global__ __launch_bounds__(256)
void transpose_w_kernel(const float* __restrict__ W, u16* __restrict__ Wt,
                        int K, int Nn) {
    __shared__ float tile[32][33];
    const int bk = blockIdx.x * 32;       // K base
    const int bn = blockIdx.y * 32;       // N base
    const int tx = threadIdx.x & 31;
    const int ty = (threadIdx.x >> 5) * 4;
#pragma unroll
    for (int r = 0; r < 4; ++r)
        tile[ty + r][tx] = W[(size_t)(bk + ty + r) * Nn + bn + tx];
    __syncthreads();
#pragma unroll
    for (int r = 0; r < 4; ++r)
        Wt[(size_t)(bn + ty + r) * K + bk + tx] = f2h(tile[tx][ty + r]);
}

// ------------------------------ GEMM ----------------------------------------
// C[M][Nn] (f32) = A[M][K] (f16) * Bt[Nn][K]^T (f16) + bias[Nn]
// 64x128x64 block tile, 8 waves (4m x 2n groups), each wave 4 16x16 tiles.
// Double-buffered LDS + register prefetch: one barrier per K stage.

#define GBM 64
#define GBN 128
#define GBK 64

__global__ __launch_bounds__(256)
void gemm_f16_kernel(const u16* __restrict__ A, const u16* __restrict__ Bt,
                     const float* __restrict__ bias, float* __restrict__ C,
                     int M, int Nn, int K) {
    __shared__ __align__(16) u16 As[2][GBM][GBK];   // 2 x 8 KB
    __shared__ __align__(16) u16 Bs[2][GBN][GBK];   // 2 x 16 KB

    const int tid      = threadIdx.x;
    const int lane     = tid & 31;
    const int wave     = tid >> 5;            // 0..7
    const int laneHalf = lane >> 4;
    const int lane16   = lane & 15;
    const int m0 = blockIdx.x * GBM;
    const int n0 = blockIdx.y * GBN;
    const int wm = (wave & 3) * 16;           // 0,16,32,48
    const int wn = (wave >> 2) * 64;          // 0 or 64

    // cooperative-load coordinates (per K stage)
    const int ar = tid >> 2, ac = (tid & 3) * 16;   // A: 64x64 u16, 32B/thread
    const int br = tid >> 1, bc = (tid & 1) * 32;   // B: 128x64 u16, 64B/thread

    const u16* aptr = A  + (size_t)(m0 + ar) * K + ac;
    const u16* bptr = Bt + (size_t)(n0 + br) * K + bc;

    uint4 pa0, pa1, pb0, pb1, pb2, pb3;
    auto fetch = [&](int k0) {
        const uint4* as = (const uint4*)(aptr + k0);
        pa0 = as[0]; pa1 = as[1];
        const uint4* bs = (const uint4*)(bptr + k0);
        pb0 = bs[0]; pb1 = bs[1]; pb2 = bs[2]; pb3 = bs[3];
    };
    auto stage = [&](int buf) {
        *(uint4*)&As[buf][ar][ac]      = pa0;
        *(uint4*)&As[buf][ar][ac + 8]  = pa1;
        *(uint4*)&Bs[buf][br][bc]      = pb0;
        *(uint4*)&Bs[buf][br][bc + 8]  = pb1;
        *(uint4*)&Bs[buf][br][bc + 16] = pb2;
        *(uint4*)&Bs[buf][br][bc + 24] = pb3;
    };

    v8f acc[4] = {};

    fetch(0);
    stage(0);
    __syncthreads();

    int buf = 0;
    for (int k0 = 0; k0 < K; k0 += GBK) {
        const bool more = (k0 + GBK) < K;
        if (more) fetch(k0 + GBK);                 // overlap with WMMA below
        if (k0 + 2 * GBK < K) {                    // warm L2 two stages ahead
            __builtin_prefetch(aptr + k0 + 2 * GBK, 0, 1);
            __builtin_prefetch(bptr + k0 + 2 * GBK, 0, 1);
        }
        const u16* arow = &As[buf][wm + lane16][0];
#pragma unroll
        for (int ks = 0; ks < 2; ++ks) {
            v16h a = load_frag(arow, ks * 32, laneHalf);
#pragma unroll
            for (int nt = 0; nt < 4; ++nt) {
                v16h b = load_frag(&Bs[buf][wn + nt * 16 + lane16][0], ks * 32, laneHalf);
                acc[nt] = wmma16(a, b, acc[nt]);
            }
        }
        if (more) {
            stage(buf ^ 1);      // other buffer: safe, last reads of it were
            __syncthreads();     // before previous barrier
            buf ^= 1;
        }
    }

    // C/D layout: VGPR r -> row r (lanes 0-15) / r+8 (lanes 16-31), col = lane16
    const int row0 = m0 + wm + laneHalf * 8;
#pragma unroll
    for (int nt = 0; nt < 4; ++nt) {
        const int col = n0 + wn + nt * 16 + lane16;
        const float bb = bias[col];
#pragma unroll
        for (int r = 0; r < 8; ++r)
            C[(size_t)(row0 + r) * Nn + col] = acc[nt][r] + bb;
    }
}

// --------------------- RoPE + head-split packing ----------------------------
// X: [B*N][D] f32 -> Y: [B][H][N][HD] f16 with interleaved-pair rotation.
__global__ __launch_bounds__(256)
void rope_pack_kernel(const float* __restrict__ X, const float* __restrict__ pe,
                      u16* __restrict__ Y, float scale) {
    int idx = blockIdx.x * 256 + threadIdx.x;     // pair index, B*N*D/2 total
    int p  = idx % (HD_ / 2);
    int hn = idx / (HD_ / 2);
    int h  = hn % H_;
    int bn = hn / H_;
    int n  = bn % N_;
    int b  = bn / N_;
    int r0 = 2 * p, r1 = 2 * p + 1;
    float x0 = X[(size_t)bn * D_ + h * HD_ + r0];
    float x1 = X[(size_t)bn * D_ + h * HD_ + r1];
    float p0 = pe[n * HD_ + r0], p1 = pe[n * HD_ + r1];
    float y0 = x0 * __cosf(p0) - x1 * __sinf(p0);
    float y1 = x1 * __cosf(p1) + x0 * __sinf(p1);
    size_t base = (((size_t)b * H_ + h) * N_ + n) * HD_;
    Y[base + r0] = f2h(y0 * scale);
    Y[base + r1] = f2h(y1 * scale);
}

// V: head-split + f16 pack + per-head transpose:
// Tmp [B*N][D] f32 -> Vht [B][H][HD][N] f16   (so attention stages V^T with b128)
__global__ __launch_bounds__(256)
void pack_v_tr_kernel(const float* __restrict__ X, u16* __restrict__ Y) {
    __shared__ u16 tile[64][68];                  // [key][dim], padded
    const int tid = threadIdx.x;
    const int kb = blockIdx.x;                    // key block (N/64)
    const int h  = blockIdx.y;
    const int b  = blockIdx.z;
    const int n0 = kb * 64;

    {   // load 64 keys x 64 dims (f32), convert, stage
        const int r = tid >> 2, c = (tid & 3) * 16;
        const float* src = X + ((size_t)b * N_ + n0 + r) * D_ + h * HD_ + c;
#pragma unroll
        for (int i = 0; i < 4; ++i) {
            float4 f = *(const float4*)(src + i * 4);
            tile[r][c + i * 4 + 0] = f2h(f.x);
            tile[r][c + i * 4 + 1] = f2h(f.y);
            tile[r][c + i * 4 + 2] = f2h(f.z);
            tile[r][c + i * 4 + 3] = f2h(f.w);
        }
    }
    __syncthreads();
    {   // write transposed rows [dim][key], vectorized 32B stores
        const int d = tid >> 2, kk = (tid & 3) * 16;
        u16 vals[16];
#pragma unroll
        for (int i = 0; i < 16; ++i) vals[i] = tile[kk + i][d];
        u16* dst = Y + (((size_t)b * H_ + h) * HD_ + d) * N_ + n0 + kk;
        *(uint4*)(dst)     = *(uint4*)&vals[0];
        *(uint4*)(dst + 8) = *(uint4*)&vals[8];
    }
}

// --------------------------- flash attention --------------------------------
// 256 threads (8 waves); wave owns 16 query rows (128 rows / block).
// Double-buffered K/V LDS; next 64-key block staged while WMMAs run.
// If the toolchain exposes them, staging uses CDNA5 async global->LDS copies
// (ASYNCcnt + s_wait_asynccnt); otherwise a register-prefetch pipeline.
__global__ __launch_bounds__(256)
void attn_kernel(const u16* __restrict__ Qh, const u16* __restrict__ Kh,
                 const u16* __restrict__ Vht, const unsigned char* __restrict__ mask,
                 u16* __restrict__ O) {
    __shared__ __align__(16) u16 Ks[2][64][HD_];   // 16 KB  [key][dim]
    __shared__ __align__(16) u16 Vs[2][HD_][64];   // 16 KB  [dim][key]
    __shared__ __align__(16) u16 Ps[8][16][64];    // 16 KB  per-wave P relayout

    const int tid      = threadIdx.x;
    const int lane     = tid & 31;
    const int wave     = tid >> 5;                // 0..7
    const int laneHalf = lane >> 4;
    const int lane16   = lane & 15;

    const int qb = blockIdx.x % (N_ / 128);
    const int bh = blockIdx.x / (N_ / 128);
    const int h  = bh % H_;
    const int b  = bh / H_;

    const size_t hoff = ((size_t)b * H_ + h) * (size_t)N_ * HD_;  // head base
    const int qrow = qb * 128 + wave * 16 + lane16;

    const u16* qbase = Qh + hoff + (size_t)qrow * HD_;
    const v16h qa0 = load_frag(qbase, 0,  laneHalf);
    const v16h qa1 = load_frag(qbase, 32, laneHalf);

    // cooperative stage coords: 8 KB per array / 256 threads = 32B per thread
    const int kr = tid >> 2, kc = (tid & 3) * 16;
    const u16* kp = Kh  + hoff + (size_t)kr * HD_ + kc;
    const u16* vp = Vht + hoff + (size_t)kr * N_  + kc;
    const unsigned char* mp = mask + b * N_;

    v8f o0 = {}, o1 = {}, o2 = {}, o3 = {};
    float rmax[8], rsum[8];
#pragma unroll
    for (int r = 0; r < 8; ++r) { rmax[r] = -3.0e38f; rsum[r] = 0.0f; }

    auto compute = [&](int buf, int j) {
        // ---- S = Q * K^T : 4 key tiles x 2 k-steps ----
        v8f s0 = {}, s1 = {}, s2 = {}, s3 = {};
        {
            const u16* k0r = &Ks[buf][ 0 + lane16][0];
            const u16* k1r = &Ks[buf][16 + lane16][0];
            const u16* k2r = &Ks[buf][32 + lane16][0];
            const u16* k3r = &Ks[buf][48 + lane16][0];
            s0 = wmma16(qa0, load_frag(k0r, 0, laneHalf), s0);
            s0 = wmma16(qa1, load_frag(k0r, 32, laneHalf), s0);
            s1 = wmma16(qa0, load_frag(k1r, 0, laneHalf), s1);
            s1 = wmma16(qa1, load_frag(k1r, 32, laneHalf), s1);
            s2 = wmma16(qa0, load_frag(k2r, 0, laneHalf), s2);
            s2 = wmma16(qa1, load_frag(k2r, 32, laneHalf), s2);
            s3 = wmma16(qa0, load_frag(k3r, 0, laneHalf), s3);
            s3 = wmma16(qa1, load_frag(k3r, 32, laneHalf), s3);
        }

        // pad mask: 4 cached byte loads per lane (L2-resident, per-head reuse)
        const float m0 = mp[j +  0 + lane16] ? -3.0e38f : 0.0f;
        const float m1 = mp[j + 16 + lane16] ? -3.0e38f : 0.0f;
        const float m2 = mp[j + 32 + lane16] ? -3.0e38f : 0.0f;
        const float m3 = mp[j + 48 + lane16] ? -3.0e38f : 0.0f;

        // ---- online softmax per row (row r in lanes 0-15, r+8 in 16-31) ----
#pragma unroll
        for (int r = 0; r < 8; ++r) {
            float e0 = s0[r] + m0, e1 = s1[r] + m1;
            float e2 = s2[r] + m2, e3 = s3[r] + m3;
            float mx = fmaxf(fmaxf(e0, e1), fmaxf(e2, e3));
#pragma unroll
            for (int off = 1; off < 16; off <<= 1)
                mx = fmaxf(mx, __shfl_xor(mx, off, 32));
            float nm    = fmaxf(rmax[r], mx);
            float alpha = __expf(rmax[r] - nm);
            float p0 = __expf(e0 - nm), p1 = __expf(e1 - nm);
            float p2 = __expf(e2 - nm), p3 = __expf(e3 - nm);
            float ps = p0 + p1 + p2 + p3;
#pragma unroll
            for (int off = 1; off < 16; off <<= 1)
                ps += __shfl_xor(ps, off, 32);
            rsum[r] = rsum[r] * alpha + ps;
            rmax[r] = nm;
            o0[r] *= alpha; o1[r] *= alpha; o2[r] *= alpha; o3[r] *= alpha;
            int prow = r + laneHalf * 8;
            Ps[wave][prow][ 0 + lane16] = f2h(p0);
            Ps[wave][prow][16 + lane16] = f2h(p1);
            Ps[wave][prow][32 + lane16] = f2h(p2);
            Ps[wave][prow][48 + lane16] = f2h(p3);
        }

        // ---- O += P * V (per-wave Ps region: no block barrier needed) ----
        v16h pa0 = load_frag(&Ps[wave][lane16][0], 0,  laneHalf);
        v16h pa1 = load_frag(&Ps[wave][lane16][0], 32, laneHalf);
        {
            const u16* v0r = &Vs[buf][ 0 + lane16][0];
            const u16* v1r = &Vs[buf][16 + lane16][0];
            const u16* v2r = &Vs[buf][32 + lane16][0];
            const u16* v3r = &Vs[buf][48 + lane16][0];
            o0 = wmma16(pa0, load_frag(v0r, 0, laneHalf), o0);
            o0 = wmma16(pa1, load_frag(v0r, 32, laneHalf), o0);
            o1 = wmma16(pa0, load_frag(v1r, 0, laneHalf), o1);
            o1 = wmma16(pa1, load_frag(v1r, 32, laneHalf), o1);
            o2 = wmma16(pa0, load_frag(v2r, 0, laneHalf), o2);
            o2 = wmma16(pa1, load_frag(v2r, 32, laneHalf), o2);
            o3 = wmma16(pa0, load_frag(v3r, 0, laneHalf), o3);
            o3 = wmma16(pa1, load_frag(v3r, 32, laneHalf), o3);
        }
    };

#if USE_ASYNC_COPY
    // --------- async DMA staging: global -> LDS with no VGPR round-trip -----
    auto issue = [&](int j, int buf) {
        async_copy_b128(kp + (size_t)j * HD_,     &Ks[buf][kr][kc]);
        async_copy_b128(kp + (size_t)j * HD_ + 8, &Ks[buf][kr][kc + 8]);
        async_copy_b128(vp + j,                   &Vs[buf][kr][kc]);
        async_copy_b128(vp + j + 8,               &Vs[buf][kr][kc + 8]);
    };
    issue(0, 0);
    int buf = 0;
    for (int j = 0; j < N_; j += 64) {
        __builtin_amdgcn_s_wait_asynccnt(0);   // my copies for `buf` done
        __syncthreads();                       // everyone's copies done
        if (j + 64 < N_) issue(j + 64, buf ^ 1);  // safe: past barrier, nobody
        compute(buf, j);                          // still reads buf^1
        buf ^= 1;
    }
#else
    // --------- register-prefetch staging (fallback) -------------------------
    uint4 rk0, rk1, rv0, rv1;
    auto fetch = [&](int j) {
        const uint4* ks = (const uint4*)(kp + (size_t)j * HD_);
        rk0 = ks[0]; rk1 = ks[1];
        const uint4* vs = (const uint4*)(vp + j);
        rv0 = vs[0]; rv1 = vs[1];
    };
    auto stage = [&](int buf) {
        *(uint4*)&Ks[buf][kr][kc]     = rk0;
        *(uint4*)&Ks[buf][kr][kc + 8] = rk1;
        *(uint4*)&Vs[buf][kr][kc]     = rv0;
        *(uint4*)&Vs[buf][kr][kc + 8] = rv1;
    };
    fetch(0);
    stage(0);
    __syncthreads();
    int buf = 0;
    for (int j = 0; j < N_; j += 64) {
        const bool more = (j + 64) < N_;
        if (more) fetch(j + 64);               // overlap with compute
        compute(buf, j);
        if (more) {
            stage(buf ^ 1);                    // other buffer: one barrier/iter
            __syncthreads();
            buf ^= 1;
        }
    }
#endif

    // ---- normalize + merge heads: O[b][n][h*64 + d] (f16) ----
#pragma unroll
    for (int r = 0; r < 8; ++r) {
        float inv = rsum[r] > 0.0f ? 1.0f / rsum[r] : 0.0f;
        int row = qb * 128 + wave * 16 + r + laneHalf * 8;
        size_t obase = ((size_t)b * N_ + row) * D_ + h * HD_;
        O[obase +  0 + lane16] = f2h(o0[r] * inv);
        O[obase + 16 + lane16] = f2h(o1[r] * inv);
        O[obase + 32 + lane16] = f2h(o2[r] * inv);
        O[obase + 48 + lane16] = f2h(o3[r] * inv);
    }
}

// ------------------------------- launch -------------------------------------

extern "C" void kernel_launch(void* const* d_in, const int* in_sizes, int n_in,
                              void* d_out, int out_size, void* d_ws, size_t ws_size,
                              hipStream_t stream) {
    (void)in_sizes; (void)n_in; (void)out_size; (void)ws_size;
    const float* q   = (const float*)d_in[0];
    const float* k   = (const float*)d_in[1];
    const float* v   = (const float*)d_in[2];
    const unsigned char* pad = (const unsigned char*)d_in[3];
    const float* pe  = (const float*)d_in[4];
    const float* Wq  = (const float*)d_in[5];
    const float* bq  = (const float*)d_in[6];
    const float* Wk  = (const float*)d_in[7];
    const float* bk  = (const float*)d_in[8];
    const float* Wv  = (const float*)d_in[9];
    const float* bv  = (const float*)d_in[10];
    const float* Wo  = (const float*)d_in[11];
    const float* bo  = (const float*)d_in[12];
    float* out = (float*)d_out;

    // workspace layout (buffers reused across stream-ordered stages), 114 MB:
    char* ws = (char*)d_ws;
    u16*   Xh  = (u16*)(ws);                           // 16 MB  activations f16
    u16*   Wt  = (u16*)(ws + (size_t)(16 << 20));      //  2 MB  W^T f16
    float* Tmp = (float*)(ws + (size_t)(18 << 20));    // 32 MB  projection f32
    u16*   Qh  = (u16*)(ws + (size_t)(50 << 20));      // 16 MB  [B][H][N][HD]
    u16*   Kh  = (u16*)(ws + (size_t)(66 << 20));      // 16 MB  [B][H][N][HD]
    u16*   Vht = (u16*)(ws + (size_t)(82 << 20));      // 16 MB  [B][H][HD][N]
    u16*   Obf = (u16*)(ws + (size_t)(98 << 20));      // 16 MB  [B*N][D]

    const int nElem = BN_ * D_;                        // 8 M
    dim3 gConv(nElem / (4 * 256));                     // 8192
    dim3 gTw(D_ / 32, D_ / 32);                        // 32 x 32
    dim3 gGemm(BN_ / GBM, D_ / GBN);                   // 128 x 8
    dim3 gRope(nElem / 2 / 256);                       // 16384
    dim3 gVtr(N_ / 64, H_, B_);                        // 32 x 16 x 4
    dim3 gAttn(B_ * H_ * (N_ / 128));                  // 1024

    // ---- Q projection + RoPE ----
    f32_to_f16_kernel<<<gConv, 256, 0, stream>>>(q, Xh, nElem);
    transpose_w_kernel<<<gTw, 256, 0, stream>>>(Wq, Wt, D_, D_);
    gemm_f16_kernel<<<gGemm, 256, 0, stream>>>(Xh, Wt, bq, Tmp, BN_, D_, D_);
    rope_pack_kernel<<<gRope, 256, 0, stream>>>(Tmp, pe, Qh, SCALE_);
    // ---- K projection + RoPE ----
    f32_to_f16_kernel<<<gConv, 256, 0, stream>>>(k, Xh, nElem);
    transpose_w_kernel<<<gTw, 256, 0, stream>>>(Wk, Wt, D_, D_);
    gemm_f16_kernel<<<gGemm, 256, 0, stream>>>(Xh, Wt, bk, Tmp, BN_, D_, D_);
    rope_pack_kernel<<<gRope, 256, 0, stream>>>(Tmp, pe, Kh, 1.0f);
    // ---- V projection (packed transposed per head) ----
    f32_to_f16_kernel<<<gConv, 256, 0, stream>>>(v, Xh, nElem);
    transpose_w_kernel<<<gTw, 256, 0, stream>>>(Wv, Wt, D_, D_);
    gemm_f16_kernel<<<gGemm, 256, 0, stream>>>(Xh, Wt, bv, Tmp, BN_, D_, D_);
    pack_v_tr_kernel<<<gVtr, 256, 0, stream>>>(Tmp, Vht);
    // ---- flash attention ----
    attn_kernel<<<gAttn, 256, 0, stream>>>(Qh, Kh, Vht, pad, Obf);
    // ---- output projection (f32 result straight to d_out) ----
    transpose_w_kernel<<<gTw, 256, 0, stream>>>(Wo, Wt, D_, D_);
    gemm_f16_kernel<<<gGemm, 256, 0, stream>>>(Obf, Wt, bo, out, BN_, D_, D_);
}
